// Mamba3Block_34926674051443
// MI455X (gfx1250) — compile-verified, matrix operand order checked by reading, not run on hardware
//
#include <hip/hip_runtime.h>
#include <hip/hip_bf16.h>

// ---------------------------------------------------------------------------
// Mamba3 block for MI455X (gfx1250, wave32, WMMA).
// Bandwidth-bound (~1.1 GB HBM traffic @ 23.3 TB/s => ~50us floor): all GEMM
// operands bf16 (f32 accumulate via v_wmma_f32_16x16x32_bf16), scan computes
// abar/u_in in-registers, and B/C are interleaved per-state (16B) so the
// sequential scan does one b128 load per lane per step.
// ---------------------------------------------------------------------------

typedef __bf16 bf16;
typedef __attribute__((ext_vector_type(16))) bf16  v16bf;
typedef __attribute__((ext_vector_type(8)))  bf16  v8bf;
typedef __attribute__((ext_vector_type(8)))  float v8f;

#define DMODEL 1024
#define DINNER 2048
#define NH     32
#define HD     64
#define NST    128     // D_STATE
#define BATCH  2
#define SEQ    2048
#define MTOK   (BATCH*SEQ)          // 4096 tokens
#define NBC    (NH*4*NST)           // 16384 used x_proj outputs per token
#define KXROW  513                  // x_proj rows per head in source weight

__device__ __forceinline__ float siluf(float v) {
    return v * (1.0f / (1.0f + __expf(-v)));
}
__device__ __forceinline__ float softplusf(float v) {
    return (v > 20.0f) ? v : log1pf(__expf(v));
}

// ---------------------------------------------------------------------------
// f32 -> bf16 convert (grid-stride)
// ---------------------------------------------------------------------------
__global__ void k_cvt_bf16(const float* __restrict__ src, bf16* __restrict__ dst, long n) {
    long i = (long)blockIdx.x * blockDim.x + threadIdx.x;
    long stride = (long)gridDim.x * blockDim.x;
    for (; i < n; i += stride) dst[i] = (bf16)src[i];
}

// Pack x_proj_w (16416x2048, 513 rows/head) -> (16384x2048 bf16, 512 rows/head)
__global__ void k_pack_xproj(const float* __restrict__ src, bf16* __restrict__ dst) {
    long n = (long)NBC * DINNER;
    long i = (long)blockIdx.x * blockDim.x + threadIdx.x;
    long stride = (long)gridDim.x * blockDim.x;
    for (; i < n; i += stride) {
        int col = (int)(i & (DINNER - 1));
        int r   = (int)(i >> 11);          // / 2048
        int h   = r >> 9;                  // / 512
        int j   = r & 511;
        dst[i] = (bf16)src[((long)(h * KXROW + j)) * DINNER + col];
    }
}

// ---------------------------------------------------------------------------
// WMMA GEMM: C[M,N] = A[M,K] (bf16 rm) x W[N,K]^T (bf16 rm). One wave/block,
// 16x64 strip, explicit register double-buffering of the K-loop so loads of
// step k+32 overlap the 4 WMMAs of step k.
// mode 0: f32 store.  mode 1: GEMM1 silu-split epilogue.
// mode 2: GEMM2 epilogue, scatter to per-head [n][Br,Bi,Cr,Ci] interleave.
// ---------------------------------------------------------------------------
__global__ void k_gemm_wmma(const bf16* __restrict__ A, const bf16* __restrict__ W,
                            float* __restrict__ Cf,
                            bf16* __restrict__ Xact, float* __restrict__ Zs,
                            int M, int N, int K, int mode) {
    const int lane = threadIdx.x & 31;
    const int half = lane >> 4;            // 0 or 1
    const int row  = lane & 15;
    const int n0 = blockIdx.x * 64;
    const int m0 = blockIdx.y * 16;

    v8f acc[4];
#pragma unroll
    for (int j = 0; j < 4; ++j)
#pragma unroll
        for (int e = 0; e < 8; ++e) acc[j][e] = 0.0f;

    const bf16* arow = A + (size_t)(m0 + row) * K + half * 8;
    const bf16* wrow[4];
#pragma unroll
    for (int j = 0; j < 4; ++j)
        wrow[j] = W + (size_t)(n0 + j * 16 + row) * K + half * 8;

    // prologue: load k=0 fragments
    v8bf alo = *(const v8bf*)(arow);
    v8bf ahi = *(const v8bf*)(arow + 16);
    v8bf blo[4], bhi[4];
#pragma unroll
    for (int j = 0; j < 4; ++j) {
        blo[j] = *(const v8bf*)(wrow[j]);
        bhi[j] = *(const v8bf*)(wrow[j] + 16);
    }

    for (int k0 = 0; k0 + 32 < K; k0 += 32) {
        const int k1 = k0 + 32;
        // prefetch A two steps ahead (global_prefetch_b8; speculative OK)
        __builtin_prefetch(arow + k1 + 32, 0, 0);
        // issue next-step loads before consuming current fragments
        v8bf nalo = *(const v8bf*)(arow + k1);
        v8bf nahi = *(const v8bf*)(arow + k1 + 16);
        v8bf nblo[4], nbhi[4];
#pragma unroll
        for (int j = 0; j < 4; ++j) {
            nblo[j] = *(const v8bf*)(wrow[j] + k1);
            nbhi[j] = *(const v8bf*)(wrow[j] + k1 + 16);
        }

        v16bf af;
#pragma unroll
        for (int i = 0; i < 8; ++i) { af[i] = alo[i]; af[i + 8] = ahi[i]; }
#pragma unroll
        for (int j = 0; j < 4; ++j) {
            v16bf bfv;
#pragma unroll
            for (int i = 0; i < 8; ++i) { bfv[i] = blo[j][i]; bfv[i + 8] = bhi[j][i]; }
            acc[j] = __builtin_amdgcn_wmma_f32_16x16x32_bf16(
                false, af, false, bfv, (short)0, acc[j], false, false);
        }

        alo = nalo; ahi = nahi;
#pragma unroll
        for (int j = 0; j < 4; ++j) { blo[j] = nblo[j]; bhi[j] = nbhi[j]; }
    }

    // epilogue K-step
    {
        v16bf af;
#pragma unroll
        for (int i = 0; i < 8; ++i) { af[i] = alo[i]; af[i + 8] = ahi[i]; }
#pragma unroll
        for (int j = 0; j < 4; ++j) {
            v16bf bfv;
#pragma unroll
            for (int i = 0; i < 8; ++i) { bfv[i] = blo[j][i]; bfv[i + 8] = bhi[j][i]; }
            acc[j] = __builtin_amdgcn_wmma_f32_16x16x32_bf16(
                false, af, false, bfv, (short)0, acc[j], false, false);
        }
    }

    // D layout: element e -> (m0 + e + half*8, n0 + j*16 + row)
#pragma unroll
    for (int j = 0; j < 4; ++j) {
        const int n = n0 + j * 16 + row;
#pragma unroll
        for (int e = 0; e < 8; ++e) {
            const int m = m0 + e + half * 8;
            float v = acc[j][e];
            if (mode == 0) {
                Cf[(size_t)m * N + n] = v;
            } else if (mode == 1) {
                float s = siluf(v);
                if (n < DINNER) Xact[(size_t)m * DINNER + n] = (bf16)s;
                else            Zs[(size_t)m * DINNER + (n - DINNER)] = s;
            } else {
                // within-head j: [s(B/C)][state][re/im] -> [state][4]
                int h = n >> 9, jj = n & 511;
                int s = jj >> 8, r = jj & 255;
                int dst = (h << 9) + ((r >> 1) << 2) + (s << 1) + (r & 1);
                Cf[(size_t)m * N + dst] = v;
            }
        }
    }
}

// ---------------------------------------------------------------------------
// Per-(token, head) prep: dt = softplus(x . dtw_h + bias_h), x_sum over
// headdim, RMS-norm of B and C in place (interleaved [n][Br,Bi,Cr,Ci]).
// ---------------------------------------------------------------------------
__device__ __forceinline__ float blockReduce128(float v, float* s) {
    int tid = threadIdx.x;
    s[tid] = v; __syncthreads();
#pragma unroll
    for (int off = 64; off > 0; off >>= 1) {
        if (tid < off) s[tid] += s[tid + off];
        __syncthreads();
    }
    float r = s[0]; __syncthreads();
    return r;
}

__global__ void k_prep(const bf16* __restrict__ Xact,
                       const float* __restrict__ dtw,   // (NH, DINNER)
                       const float* __restrict__ dtb,   // (NH)
                       float* __restrict__ BC,          // (MTOK, NBC) in/out
                       const float* __restrict__ wB,    // (256)
                       const float* __restrict__ wC,    // (256)
                       float* __restrict__ dtArr,       // (MTOK, NH)
                       float* __restrict__ dtxsArr) {   // (MTOK, NH)
    __shared__ float s[128];
    const int blk = blockIdx.x;
    const int m = blk / NH;
    const int h = blk % NH;
    const int tid = threadIdx.x;

    // dt dot product over K=2048: 16 contiguous bf16 per thread
    float pd = 0.0f;
    const bf16* xr = Xact + (size_t)m * DINNER;
    const float* wr = dtw + (size_t)h * DINNER;
    {
        const int kb = tid * 16;
        v8bf x0 = *(const v8bf*)(xr + kb);
        v8bf x1 = *(const v8bf*)(xr + kb + 8);
#pragma unroll
        for (int i = 0; i < 8; ++i) pd += (float)x0[i] * wr[kb + i];
#pragma unroll
        for (int i = 0; i < 8; ++i) pd += (float)x1[i] * wr[kb + 8 + i];
    }
    float dot = blockReduce128(pd, s);

    // x_sum over headdim (64)
    float px = (tid < HD) ? (float)xr[h * HD + tid] : 0.0f;
    float xs = blockReduce128(px, s);

    // RMS norm: thread owns state n = tid, 16B interleaved record
    float* base = BC + (size_t)m * NBC + (size_t)h * (4 * NST);
    float4 v = *(float4*)(base + 4 * tid);
    float ssB = blockReduce128(v.x * v.x + v.y * v.y, s);
    float ssC = blockReduce128(v.z * v.z + v.w * v.w, s);
    float rB = rsqrtf(ssB * (1.0f / 256.0f) + 1e-6f);
    float rC = rsqrtf(ssC * (1.0f / 256.0f) + 1e-6f);
    v.x *= rB * wB[2 * tid];
    v.y *= rB * wB[2 * tid + 1];
    v.z *= rC * wC[2 * tid];
    v.w *= rC * wC[2 * tid + 1];
    *(float4*)(base + 4 * tid) = v;

    if (tid == 0) {
        float dtv = softplusf(dot + dtb[h]);
        dtArr[(size_t)m * NH + h]   = dtv;
        dtxsArr[(size_t)m * NH + h] = dtv * xs;
    }
}

// ---------------------------------------------------------------------------
// Sequential complex scan. One 128-thread block per (b,h); lane owns state n.
// One b128 load per lane per step (Br,Bi,Cr,Ci). abar/u_in in registers.
// ---------------------------------------------------------------------------
__global__ void k_scan(const float* __restrict__ BC,
                       const float* __restrict__ Ap,     // (NH, NST, 2)
                       const float* __restrict__ dtArr,
                       const float* __restrict__ dtxsArr,
                       float* __restrict__ yred) {       // (MTOK, NH)
    __shared__ float wsum[4];
    const int blk = blockIdx.x;
    const int b = blk / NH;
    const int h = blk % NH;
    const int n = threadIdx.x;          // 0..127

    const float Ar = Ap[((size_t)h * NST + n) * 2 + 0];
    const float Ai = Ap[((size_t)h * NST + n) * 2 + 1];

    float hr = 0.0f, hi = 0.0f;

    for (int t = 0; t < SEQ; ++t) {
        const int m = b * SEQ + t;
        const float dtv = dtArr[(size_t)m * NH + h];
        const float dxs = dtxsArr[(size_t)m * NH + h];
        const float4 bc = *(const float4*)(BC + (size_t)m * NBC
                                           + (size_t)h * (4 * NST) + 4 * n);

        // zt = A*dt/2 ; den = 1-zt ; abar = (1+zt)/den ; s = dxs/den
        const float ztr = 0.5f * dtv * Ar;
        const float zti = 0.5f * dtv * Ai;
        const float dr = 1.0f - ztr, di = -zti;
        const float inv = 1.0f / (dr * dr + di * di);
        const float pr = 1.0f + ztr;
        const float ar = (pr * dr + zti * di) * inv;
        const float ai = (zti * dr - pr * di) * inv;
        const float sr =  dxs * dr * inv;
        const float si = -dxs * di * inv;
        const float ur = bc.x * sr - bc.y * si;
        const float ui = bc.x * si + bc.y * sr;

        const float hr2 = ar * hr - ai * hi + ur;
        hi = ar * hi + ai * hr + ui;
        hr = hr2;

        float p = bc.z * hr - bc.w * hi;    // Re(C * h)
#pragma unroll
        for (int off = 16; off > 0; off >>= 1) p += __shfl_xor(p, off, 32);
        if ((n & 31) == 0) wsum[n >> 5] = p;
        __syncthreads();
        if (n == 0) yred[(size_t)m * NH + h] = wsum[0] + wsum[1] + wsum[2] + wsum[3];
        __syncthreads();
    }
}

// ---------------------------------------------------------------------------
// Gate: y = silu(x) * (y_red + D) * silu(z) -> bf16 for the final GEMM
// ---------------------------------------------------------------------------
__global__ void k_gate(const bf16* __restrict__ Xact,
                       const float* __restrict__ yred,
                       const float* __restrict__ Dp,
                       const float* __restrict__ Zs,
                       bf16* __restrict__ Ybf) {
    long n = (long)MTOK * DINNER;
    long i = (long)blockIdx.x * blockDim.x + threadIdx.x;
    long stride = (long)gridDim.x * blockDim.x;
    for (; i < n; i += stride) {
        int c = (int)(i & (DINNER - 1));
        long m = i >> 11;
        int hh = c >> 6;
        float v = (float)Xact[i] * (yred[m * NH + hh] + Dp[hh]) * Zs[i];
        Ybf[i] = (bf16)v;
    }
}

// ---------------------------------------------------------------------------
// Launch
// ---------------------------------------------------------------------------
extern "C" void kernel_launch(void* const* d_in, const int* in_sizes, int n_in,
                              void* d_out, int out_size, void* d_ws, size_t ws_size,
                              hipStream_t stream) {
    (void)in_sizes; (void)n_in; (void)out_size; (void)ws_size;

    const float* u      = (const float*)d_in[0];   // (B,L,1024)
    const float* w_in   = (const float*)d_in[1];   // (4096,1024)
    const float* dtw    = (const float*)d_in[2];   // (32,2048)
    const float* dtb    = (const float*)d_in[3];   // (32)
    const float* w_xprj = (const float*)d_in[4];   // (16416,2048)
    const float* Ap     = (const float*)d_in[5];   // (32,128,2)
    const float* Dp     = (const float*)d_in[6];   // (32)
    const float* wB     = (const float*)d_in[7];   // (256)
    const float* wC     = (const float*)d_in[8];   // (256)
    const float* w_out  = (const float*)d_in[9];   // (1024,2048)
    float* out          = (float*)d_out;           // (B,L,1024)

    // workspace carve-up (256B aligned)
    char* ws = (char*)d_ws;
    size_t off = 0;
    auto alloc = [&](size_t bytes) {
        void* p = ws + off;
        off += (bytes + 255) & ~(size_t)255;
        return p;
    };
    bf16*  W1   = (bf16*)alloc((size_t)4096 * 1024 * 2);       // in_proj  bf16
    bf16*  W2   = (bf16*)alloc((size_t)NBC * DINNER * 2);      // packed x_proj bf16
    bf16*  W3   = (bf16*)alloc((size_t)1024 * 2048 * 2);       // out_proj bf16
    bf16*  Ubf  = (bf16*)alloc((size_t)MTOK * DMODEL * 2);
    bf16*  Xact = (bf16*)alloc((size_t)MTOK * DINNER * 2);     // silu(x) bf16
    float* Zs   = (float*)alloc((size_t)MTOK * DINNER * 4);    // silu(z) f32
    float* BC   = (float*)alloc((size_t)MTOK * NBC * 4);       // interleaved B/C
    float* dtA  = (float*)alloc((size_t)MTOK * NH * 4);
    float* dtxs = (float*)alloc((size_t)MTOK * NH * 4);
    float* yred = (float*)alloc((size_t)MTOK * NH * 4);
    bf16*  Ybf  = (bf16*)alloc((size_t)MTOK * DINNER * 2);

    // 1) converts / packing
    k_cvt_bf16<<<4096, 256, 0, stream>>>(u,     Ubf, (long)MTOK * DMODEL);
    k_cvt_bf16<<<4096, 256, 0, stream>>>(w_in,  W1,  (long)4096 * 1024);
    k_cvt_bf16<<<4096, 256, 0, stream>>>(w_out, W3,  (long)1024 * 2048);
    k_pack_xproj<<<8192, 256, 0, stream>>>(w_xprj, W2);

    // 2) GEMM1: xz = u @ W1^T, fused silu split epilogue
    {
        dim3 g((2 * DINNER) / 64, MTOK / 16);
        k_gemm_wmma<<<g, 32, 0, stream>>>(Ubf, W1, nullptr, Xact, Zs,
                                          MTOK, 2 * DINNER, DMODEL, 1);
    }

    // 3) GEMM2: BC = silu(x) @ W2^T, interleaved scatter epilogue
    {
        dim3 g(NBC / 64, MTOK / 16);
        k_gemm_wmma<<<g, 32, 0, stream>>>(Xact, W2, BC, nullptr, nullptr,
                                          MTOK, NBC, DINNER, 2);
    }

    // 4) dt / x_sum / RMS-norm of B,C
    k_prep<<<MTOK * NH, 128, 0, stream>>>(Xact, dtw, dtb, BC, wB, wC, dtA, dtxs);

    // 5) sequential complex scan -> y_red
    k_scan<<<BATCH * NH, 128, 0, stream>>>(BC, Ap, dtA, dtxs, yred);

    // 6) gate -> Ybf
    k_gate<<<8192, 256, 0, stream>>>(Xact, yred, Dp, Zs, Ybf);

    // 7) GEMM3: out = y @ W3^T
    {
        dim3 g(DMODEL / 64, MTOK / 16);
        k_gemm_wmma<<<g, 32, 0, stream>>>(Ybf, W3, out, nullptr, nullptr,
                                          MTOK, DMODEL, DINNER, 0);
    }
}
